// PointGNN_63316407878452
// MI455X (gfx1250) — matrix operand order repaired and checked by reading, not compiled, not run on hardware
//
#include <hip/hip_runtime.h>
#include <cstdint>
#include <cstddef>

// ---------------------------------------------------------------------------
// Types for CDNA5 WMMA bf16: D(f32 16x16) = A(16x32 bf16) * B(32x16 bf16) + C
// ---------------------------------------------------------------------------
typedef __attribute__((ext_vector_type(16))) __bf16 v16bf;
typedef __attribute__((ext_vector_type(8)))  float  v8f;

struct Frag32B { uint4 a, b; };   // 32 bytes == one v16bf fragment

__device__ __forceinline__ float bf2f(unsigned short h) {
  return __uint_as_float(((unsigned)h) << 16);
}
__device__ __forceinline__ unsigned short f2bf(float f) {
  unsigned u = __float_as_uint(f);
  u += 0x7FFFu + ((u >> 16) & 1u);          // round-to-nearest-even
  return (unsigned short)(u >> 16);
}

// ---------------------------------------------------------------------------
// Weight packing: f32 W[K,N] -> bf16 B-matrix fragments.
// Per 16x32 tile (256 dwords) layout is [vpair][lane][lo/hi-dword]:
//   dst[tile*256 + vp*64 + lane*2 + (v&1)]  with v = 2*vp + (v&1)
// so a B-fragment load is 4x ds_load_b64 where 32 lanes cover 256 consecutive
// bytes -> one dword per LDS bank, conflict-free.
// Fragment dword v holds the bf16 pair (k, k+1):
//   n = nt*16 + (lane&15);  k = kt*32 + ((lane&16)?16:0) + 2*v
// ---------------------------------------------------------------------------
__global__ void pack_w(const float* __restrict__ W, int Ktrue, int Ntrue,
                       int NT, uint32_t* __restrict__ dst, int total) {
  int i = blockIdx.x * 256 + threadIdx.x;
  if (i >= total) return;
  int r    = i & 255;
  int tile = i >> 8;              // kt*NT + nt
  int vp   = r >> 6;              // 0..3
  int lane = (r >> 1) & 31;
  int v    = vp * 2 + (r & 1);
  int nt   = tile % NT;
  int kt   = tile / NT;
  int n = nt * 16 + (lane & 15);
  int k = kt * 32 + ((lane & 16) ? 16 : 0) + 2 * v;
  unsigned lo = 0, hi = 0;
  if (n < Ntrue) {
    if (k     < Ktrue) lo = f2bf(W[(size_t)k       * Ntrue + n]);
    if (k + 1 < Ktrue) hi = f2bf(W[(size_t)(k + 1) * Ntrue + n]);
  }
  dst[i] = lo | (hi << 16);
}

// f32 [rows, cin] -> bf16 [rows, ldy] zero-padded
__global__ void cvt_pad_bf(const float* __restrict__ x, unsigned short* __restrict__ y,
                           int rows, int cin, int ldy) {
  int i = blockIdx.x * 256 + threadIdx.x;
  int r = i / ldy, c = i % ldy;
  if (r >= rows) return;
  y[i] = (c < cin) ? f2bf(x[(size_t)r * cin + c]) : (unsigned short)0;
}

__global__ void cvt_f32_bf(const float* __restrict__ x, unsigned short* __restrict__ y, int n) {
  int i = blockIdx.x * 256 + threadIdx.x;
  if (i < n) y[i] = f2bf(x[i]);
}

__global__ void zero_f32(float* __restrict__ p, int n) {
  int i = blockIdx.x * 256 + threadIdx.x;
  if (i < n) p[i] = 0.0f;
}

// contiguous 8-way segment max over keypoint features (bf16 [NV*8,320] -> [NV,320])
__global__ void segmax8(const unsigned short* __restrict__ kp,
                        unsigned short* __restrict__ out, int NV) {
  int i = blockIdx.x * 256 + threadIdx.x;
  int v = i / 320, c = i % 320;
  if (v >= NV) return;
  size_t base = (size_t)v * 8 * 320 + c;
  float m = bf2f(kp[base]);
  #pragma unroll
  for (int r = 1; r < 8; ++r) m = fmaxf(m, bf2f(kp[base + (size_t)r * 320]));
  out[(size_t)v * 320 + c] = f2bf(m);
}

// tmp[e, 0:3) = pos[nbr]-pos[src]-delta ; tmp[e, 3:303) = s[nbr] ; rest = 0
__global__ void edge_concat(const float* __restrict__ pos,
                            const int* __restrict__ src, const int* __restrict__ nbr,
                            const unsigned short* __restrict__ delta,   // bf16 [E,32]
                            const unsigned short* __restrict__ s,       // bf16 [NV,320]
                            unsigned short* __restrict__ tmp,           // bf16 [E,320]
                            int E) {
  int g = blockIdx.x * 256 + threadIdx.x;
  int e = g >> 5, lane = g & 31;
  if (e >= E) return;
  int a = src[e], b = nbr[e];
  for (int c = lane; c < 320; c += 32) {
    unsigned short o;
    if (c < 3) {
      float val = pos[(size_t)b * 3 + c] - pos[(size_t)a * 3 + c]
                  - bf2f(delta[(size_t)e * 32 + c]);
      o = f2bf(val);
    } else if (c < 303) {
      o = s[(size_t)b * 320 + (c - 3)];
    } else {
      o = 0;
    }
    tmp[(size_t)e * 320 + c] = o;
  }
}

// ---------------------------------------------------------------------------
// Fused block layer:  Y = ReLU(InstanceNorm(X @ W + b))  via WMMA bf16.
//   - X bf16 [M, KT*32] (optionally gathered through row_map)
//   - packed weights resident in LDS (<=200KB, conflict-free b64 frag loads)
//   - per-wave: one 16-row stripe, all NT 16-col tiles held in VGPRs
//   - row stats via 16-lane butterfly reductions (C layout: halves = row+0/+8)
//   - optional residual add, f32 output slice, or fused atomic segment-max
// ---------------------------------------------------------------------------
template<int KT, int NT>
__global__ __launch_bounds__(128)
__attribute__((amdgpu_waves_per_eu(1)))
void gemm_block(const unsigned short* __restrict__ X, int ldx,
                const int* __restrict__ rmap,
                const uint32_t* __restrict__ wpk,
                const float* __restrict__ bias, int Ntrue,
                unsigned short* __restrict__ Y, int ldy,
                const unsigned short* __restrict__ res,
                float* __restrict__ outf, int ldo, int coff, int ocols,
                const int* __restrict__ amap, float* __restrict__ aout,
                int M) {
  __shared__ uint32_t wl[KT * NT * 256];
  for (int i = threadIdx.x; i < KT * NT * 256; i += 128) wl[i] = wpk[i];
  __syncthreads();

  const int lane = threadIdx.x & 31;
  const int wave = threadIdx.x >> 5;
  const int l15  = lane & 15;
  const int hi   = (lane >> 4) & 1;
  const int Mt   = (M + 15) >> 4;
  const int tstep = gridDim.x * 4;

  // hoist per-column bias + validity out of the tile loop
  float bvals[NT];
  bool  bvalid[NT];
  #pragma unroll
  for (int nt = 0; nt < NT; ++nt) {
    int n = nt * 16 + l15;
    bvalid[nt] = n < Ntrue;
    bvals[nt]  = bvalid[nt] ? bias[n] : 0.f;
  }
  const float inv = 1.0f / (float)Ntrue;

  for (int tile = blockIdx.x * 4 + wave; tile < Mt; tile += tstep) {
    int m0 = tile * 16 + l15;
    int mr = (m0 < M) ? m0 : (M - 1);
    int row = rmap ? rmap[mr] : mr;
    const unsigned short* xr = X + (size_t)((unsigned)row * (unsigned)ldx) + hi * 16;

    // prefetch next grid-stride tile's A source (global_prefetch_b8)
    int m0n = (tile + tstep) * 16 + l15;
    if (m0n < M) {
      if (rmap) __builtin_prefetch(&rmap[m0n], 0, 1);
      else      __builtin_prefetch(X + (size_t)((unsigned)m0n * (unsigned)ldx), 0, 1);
    }

    v8f acc[NT];
    v8f zero = {0.f, 0.f, 0.f, 0.f, 0.f, 0.f, 0.f, 0.f};
    #pragma unroll
    for (int nt = 0; nt < NT; ++nt) acc[nt] = zero;

    #pragma unroll
    for (int kt = 0; kt < KT; ++kt) {
      Frag32B fa;
      fa.a = *(const uint4*)(xr + kt * 32);
      fa.b = *(const uint4*)(xr + kt * 32 + 8);
      v16bf av = __builtin_bit_cast(v16bf, fa);
      const uint32_t* wb = wl + kt * NT * 256 + lane * 2;
      #pragma unroll
      for (int nt = 0; nt < NT; ++nt) {
        const uint32_t* p = wb + nt * 256;
        uint2 q0 = *(const uint2*)(p);          // v0,v1
        uint2 q1 = *(const uint2*)(p + 64);     // v2,v3
        uint2 q2 = *(const uint2*)(p + 128);    // v4,v5
        uint2 q3 = *(const uint2*)(p + 192);    // v6,v7
        Frag32B fb;
        fb.a.x = q0.x; fb.a.y = q0.y; fb.a.z = q1.x; fb.a.w = q1.y;
        fb.b.x = q2.x; fb.b.y = q2.y; fb.b.z = q3.x; fb.b.w = q3.y;
        v16bf bv = __builtin_bit_cast(v16bf, fb);
        acc[nt] = __builtin_amdgcn_wmma_f32_16x16x32_bf16(
            false, av, false, bv, (short)0, acc[nt], false, false);
      }
    }

    // ---- per-row InstanceNorm stats (biased var, eps=1e-5) ----
    float sum[8] = {}, ss[8] = {};
    #pragma unroll
    for (int nt = 0; nt < NT; ++nt) {
      #pragma unroll
      for (int v = 0; v < 8; ++v) {
        float val = bvalid[nt] ? (acc[nt][v] + bvals[nt]) : 0.f;
        sum[v] += val;
        ss[v]  += val * val;
      }
    }
    #pragma unroll
    for (int v = 0; v < 8; ++v) {
      #pragma unroll
      for (int msk = 1; msk < 16; msk <<= 1) {
        sum[v] += __shfl_xor(sum[v], msk, 16);
        ss[v]  += __shfl_xor(ss[v],  msk, 16);
      }
    }
    float mean[8], rstd[8];
    #pragma unroll
    for (int v = 0; v < 8; ++v) {
      mean[v] = sum[v] * inv;
      float var = ss[v] * inv - mean[v] * mean[v];
      rstd[v] = rsqrtf(var + 1e-5f);
    }

    // ---- epilogue: norm + relu (+residual / f32-out / atomic segment-max) ----
    #pragma unroll
    for (int nt = 0; nt < NT; ++nt) {
      int n = nt * 16 + l15;
      #pragma unroll
      for (int v = 0; v < 8; ++v) {
        int mrow = tile * 16 + v + hi * 8;
        if (mrow >= M) continue;
        float val = 0.f;
        if (bvalid[nt]) {
          val = (acc[nt][v] + bvals[nt] - mean[v]) * rstd[v];
          val = val > 0.f ? val : 0.f;
        }
        float outv = val;
        unsigned roff = (unsigned)mrow * (unsigned)ldy + (unsigned)n;
        if (res) outv += bf2f(res[roff]);                     // pad cols hold 0
        if (Y)   Y[roff] = f2bf(outv);
        if (outf && bvalid[nt] && n < ocols)
          outf[(size_t)((unsigned)mrow * (unsigned)ldo) + coff + n] = outv;
        if (aout && bvalid[nt]) {                             // val >= 0: u32-bit max == float max
          atomicMax((unsigned int*)&aout[(size_t)((unsigned)amap[mrow] * (unsigned)(NT * 16)) + n],
                    __float_as_uint(val));
        }
      }
    }
  }
}

// host-side launcher
template<int KT, int NT>
static inline void run_gemm(hipStream_t st, const unsigned short* X, int ldx,
                            const int* rmap, const uint32_t* pk, const float* bias,
                            int Ntrue, unsigned short* Y, int ldy,
                            const unsigned short* res,
                            float* outf, int ldo, int coff, int ocols,
                            const int* amap, float* aout, int M) {
  int Mt = (M + 15) / 16;
  int g = (Mt + 3) / 4;
  if (g > 1280) g = 1280;
  if (g < 1) g = 1;
  gemm_block<KT, NT><<<g, 128, 0, st>>>(X, ldx, rmap, pk, bias, Ntrue, Y, ldy, res,
                                        outf, ldo, coff, ocols, amap, aout, M);
}

// ---------------------------------------------------------------------------
// Orchestration.  Input order (insertion order of setup_inputs / params):
//  0 key_points, 1 pos,
//  2..9  init (W,b)x4, 10..13 init_aggr (W,b)x2,
//  14+12t: h(W,b)x2, f(W,b)x2, g(W,b)x2  for t=0..2,
//  50..53 cls (W,b)x2, 54+6c: loc (W,b)x3 for c=0..3,
//  n_in-2 key_points_lookup, n_in-1 edge_index[2,E]
// ---------------------------------------------------------------------------
extern "C" void kernel_launch(void* const* d_in, const int* in_sizes, int n_in,
                              void* d_out, int out_size, void* d_ws, size_t ws_size,
                              hipStream_t stream) {
  (void)out_size; (void)ws_size;
  const int NV = in_sizes[1] / 3;            // 16384 vertices
  const int NK = in_sizes[0] / 4;            // 131072 keypoints
  const int E  = in_sizes[n_in - 1] / 2;     // 196608 edges
  const int*   edge = (const int*)d_in[n_in - 1];
  const int*   esrc = edge;
  const int*   enbr = edge + E;
  const float* pos  = (const float*)d_in[1];

  char* base = (char*)d_ws;
  size_t off = 0;
  auto carve = [&](size_t bytes) -> char* {
    char* p = base + off;
    off = (off + bytes + 255) & ~(size_t)255;
    return p;
  };

  struct PL { const uint32_t* pk; const float* b; int N; };
  auto pack = [&](int widx, int K, int N, int KT, int NT) -> PL {
    size_t cnt = (size_t)KT * NT * 256;
    uint32_t* dst = (uint32_t*)carve(cnt * 4);
    pack_w<<<(int)((cnt + 255) / 256), 256, 0, stream>>>(
        (const float*)d_in[widx], K, N, NT, dst, (int)cnt);
    return PL{dst, (const float*)d_in[widx + 1], N};
  };

  PL init0 = pack(2, 4,   32, 1, 2);
  PL init1 = pack(4, 32,  64, 1, 4);
  PL init2 = pack(6, 64, 128, 2, 8);
  PL init3 = pack(8, 128, 300, 4, 20);
  PL ia0   = pack(10, 300, 300, 10, 20);
  PL ia1   = pack(12, 300, 300, 10, 20);
  PL h0[3], h1[3], f0[3], f1[3], g0[3], g1[3];
  for (int t = 0; t < 3; ++t) {
    int b = 14 + t * 12;
    h0[t] = pack(b + 0, 300, 64, 10, 4);
    h1[t] = pack(b + 2, 64,   3, 2, 2);
    f0[t] = pack(b + 4, 303, 300, 10, 20);
    f1[t] = pack(b + 6, 300, 300, 10, 20);
    g0[t] = pack(b + 8, 300, 300, 10, 20);
    g1[t] = pack(b + 10, 300, 300, 10, 20);
  }
  PL cls0 = pack(50, 300, 64, 10, 4);
  PL cls1 = pack(52, 64,   4, 2, 2);
  PL loc0[4], loc1[4], loc2[4];
  for (int c = 0; c < 4; ++c) {
    int b = 54 + c * 6;
    loc0[c] = pack(b + 0, 300, 64, 10, 4);
    loc1[c] = pack(b + 2, 64,  64, 2, 4);
    loc2[c] = pack(b + 4, 64,   7, 2, 2);
  }

  // activation buffers
  unsigned short* kpx  = (unsigned short*)carve((size_t)NK * 32 * 2);
  unsigned short* R1   = (unsigned short*)carve((size_t)E * 320 * 2);
  unsigned short* R2   = (unsigned short*)carve((size_t)E * 320 * 2);
  unsigned short* s0   = (unsigned short*)carve((size_t)NV * 320 * 2);
  unsigned short* s1   = (unsigned short*)carve((size_t)NV * 320 * 2);
  unsigned short* nagg = (unsigned short*)carve((size_t)NV * 320 * 2);
  unsigned short* ngt  = (unsigned short*)carve((size_t)NV * 320 * 2);
  float*          aggF = (float*)carve((size_t)NV * 320 * 4);

  // ---- initial vertex state ----
  cvt_pad_bf<<<(int)(((size_t)NK * 32 + 255) / 256), 256, 0, stream>>>(
      (const float*)d_in[0], kpx, NK, 4, 32);
  run_gemm<1, 2>(stream, kpx, 32, nullptr, init0.pk, init0.b, 32,  R1, 32,  nullptr, nullptr, 0, 0, 0, nullptr, nullptr, NK);
  run_gemm<1, 4>(stream, R1, 32,  nullptr, init1.pk, init1.b, 64,  R2, 64,  nullptr, nullptr, 0, 0, 0, nullptr, nullptr, NK);
  run_gemm<2, 8>(stream, R2, 64,  nullptr, init2.pk, init2.b, 128, R1, 128, nullptr, nullptr, 0, 0, 0, nullptr, nullptr, NK);
  run_gemm<4, 20>(stream, R1, 128, nullptr, init3.pk, init3.b, 300, R2, 320, nullptr, nullptr, 0, 0, 0, nullptr, nullptr, NK);
  segmax8<<<(int)(((size_t)NV * 320 + 255) / 256), 256, 0, stream>>>(R2, nagg, NV);
  run_gemm<10, 20>(stream, nagg, 320, nullptr, ia0.pk, ia0.b, 300, ngt, 320, nullptr, nullptr, 0, 0, 0, nullptr, nullptr, NV);
  run_gemm<10, 20>(stream, ngt, 320, nullptr, ia1.pk, ia1.b, 300, s0,  320, nullptr, nullptr, 0, 0, 0, nullptr, nullptr, NV);

  // ---- message-passing iterations ----
  for (int t = 0; t < 3; ++t) {
    unsigned short* cur = (t & 1) ? s1 : s0;
    unsigned short* nxt = (t & 1) ? s0 : s1;
    // h: delta_x = MLP(s[src]); gather fused via row_map
    run_gemm<10, 4>(stream, cur, 320, esrc, h0[t].pk, h0[t].b, 64, R1, 64, nullptr, nullptr, 0, 0, 0, nullptr, nullptr, E);
    run_gemm<2, 2>(stream, R1, 64, nullptr, h1[t].pk, h1[t].b, 3,  R2, 32, nullptr, nullptr, 0, 0, 0, nullptr, nullptr, E);
    // tmp = [pos[nbr]-pos[src]-delta, s[nbr]]
    edge_concat<<<(int)(((size_t)E * 32 + 255) / 256), 256, 0, stream>>>(
        pos, esrc, enbr, R2, cur, R1, E);
    zero_f32<<<(int)(((size_t)NV * 320 + 255) / 256), 256, 0, stream>>>(aggF, NV * 320);
    // f: e_ij, second layer fuses atomic segment-max at src
    run_gemm<10, 20>(stream, R1, 320, nullptr, f0[t].pk, f0[t].b, 300, R2, 320, nullptr, nullptr, 0, 0, 0, nullptr, nullptr, E);
    run_gemm<10, 20>(stream, R2, 320, nullptr, f1[t].pk, f1[t].b, 300, nullptr, 320, nullptr, nullptr, 0, 0, 0, esrc, aggF, E);
    cvt_f32_bf<<<(int)(((size_t)NV * 320 + 255) / 256), 256, 0, stream>>>(aggF, nagg, NV * 320);
    // g + residual
    run_gemm<10, 20>(stream, nagg, 320, nullptr, g0[t].pk, g0[t].b, 300, ngt, 320, nullptr, nullptr, 0, 0, 0, nullptr, nullptr, NV);
    run_gemm<10, 20>(stream, ngt, 320, nullptr, g1[t].pk, g1[t].b, 300, nxt, 320, cur, nullptr, 0, 0, 0, nullptr, nullptr, NV);
  }
  unsigned short* sfin = s1;  // after t = 0,1,2

  // ---- heads: cls -> d_out[:, 0:4], loc -> d_out offset NV*4, [NV, 28] ----
  float* out_cls = (float*)d_out;
  float* out_reg = (float*)d_out + (size_t)NV * 4;
  run_gemm<10, 4>(stream, sfin, 320, nullptr, cls0.pk, cls0.b, 64, ngt, 64, nullptr, nullptr, 0, 0, 0, nullptr, nullptr, NV);
  run_gemm<2, 2>(stream, ngt, 64, nullptr, cls1.pk, cls1.b, 4, nullptr, 32, nullptr, out_cls, 4, 0, 4, nullptr, nullptr, NV);
  for (int c = 0; c < 4; ++c) {
    run_gemm<10, 4>(stream, sfin, 320, nullptr, loc0[c].pk, loc0[c].b, 64, ngt, 64, nullptr, nullptr, 0, 0, 0, nullptr, nullptr, NV);
    run_gemm<2, 4>(stream, ngt, 64, nullptr, loc1[c].pk, loc1[c].b, 64, nagg, 64, nullptr, nullptr, 0, 0, 0, nullptr, nullptr, NV);
    run_gemm<2, 2>(stream, nagg, 64, nullptr, loc2[c].pk, loc2[c].b, 7, nullptr, 32, nullptr, out_reg, 28, 7 * c, 7, nullptr, nullptr, NV);
  }
}
